// DecoderBlock_75110388072826
// MI455X (gfx1250) — matrix operand order repaired and checked by reading, not compiled, not run on hardware
//
#include <hip/hip_runtime.h>

#define D_MODEL 2048
#define NHEADS  16
#define HDIM    128
#define SEQ     2048
#define BATCH   2
#define BT      (BATCH * SEQ)   // 4096 rows
#define DFF     8192            // D * MULT

typedef __attribute__((ext_vector_type(16))) __bf16 bf16x16;
typedef __attribute__((ext_vector_type(8)))  float  f32x8;

union Frag16 { bf16x16 bf; unsigned int u[8]; };

__device__ __forceinline__ unsigned short f2bf(float f) {
  unsigned int u = __float_as_uint(f);
  u += 0x7FFFu + ((u >> 16) & 1u);   // round-to-nearest-even
  return (unsigned short)(u >> 16);
}

__device__ __forceinline__ f32x8 wmma_bf16(const Frag16& a, const Frag16& b, f32x8 c) {
  return __builtin_amdgcn_wmma_f32_16x16x32_bf16(false, a.bf, false, b.bf,
                                                 (short)0, c, false, false);
}

__device__ __forceinline__ float gelu_exact(float x) {
  return 0.5f * x * (1.0f + erff(x * 0.70710678118654752f));
}

// ---- CDNA5 async global->LDS copy (ASYNCcnt path, no VGPR round-trip) ----
__device__ __forceinline__ void async_copy_b128(const void* gptr, void* lds_ptr) {
  unsigned lds = (unsigned)(unsigned long long)lds_ptr;  // low 32 bits = LDS offset
  asm volatile("global_load_async_to_lds_b128 %0, %1, off"
               :: "v"(lds), "v"(gptr) : "memory");
}
__device__ __forceinline__ void wait_async0() {
  asm volatile("s_wait_asynccnt 0x0" ::: "memory");
}
__device__ __forceinline__ void wait_async4() {
  asm volatile("s_wait_asynccnt 0x4" ::: "memory");
}
__device__ __forceinline__ void wait_async6() {
  asm volatile("s_wait_asynccnt 0x6" ::: "memory");
}

// ---------------------------------------------------------------- converts
__global__ __launch_bounds__(256) void f32_to_bf16_kernel(
    const float* __restrict__ in, unsigned short* __restrict__ out, size_t n) {
  size_t i = (size_t)blockIdx.x * 256 + threadIdx.x;
  size_t stride = (size_t)gridDim.x * 256;
  for (; i < n; i += stride) out[i] = f2bf(in[i]);
}

// ---------------------------------------------------------------- rmsnorm
__global__ __launch_bounds__(256) void rmsnorm_bf16_kernel(
    const float* __restrict__ x, const float* __restrict__ g,
    unsigned short* __restrict__ out) {
  __shared__ float red[256];
  const int tid = threadIdx.x;
  const size_t base = (size_t)blockIdx.x * D_MODEL;
  float s = 0.f;
  for (int i = tid; i < D_MODEL; i += 256) { float v = x[base + i]; s += v * v; }
  red[tid] = s; __syncthreads();
  for (int st = 128; st > 0; st >>= 1) {
    if (tid < st) red[tid] += red[tid + st];
    __syncthreads();
  }
  const float inv = rsqrtf(red[0] * (1.0f / D_MODEL) + 1e-6f);
  for (int i = tid; i < D_MODEL; i += 256)
    out[base + i] = f2bf(g[i] * x[base + i] * inv);
}

// ---------------------------------------------------------------- GEMM: C = A @ B^T
// A: [M,K] bf16 row-major; B: [N,K] bf16 row-major. 64x64 block tile, BK=64,
// double-buffered LDS fed by async copies.
// mode 0: Cf = acc ; mode 1: Cf = acc + Res ; mode 2: Cbf = bf16(acc)
__global__ __launch_bounds__(256) void gemm_bf16_kernel(
    const unsigned short* __restrict__ A, const unsigned short* __restrict__ B,
    const float* __restrict__ Res, float* __restrict__ Cf,
    unsigned short* __restrict__ Cbf, int M, int N, int K, int mode) {
  __shared__ __align__(16) unsigned short Asub[2][64][64];
  __shared__ __align__(16) unsigned short Bsub[2][64][64];

  const int tid = threadIdx.x;
  const int lane = tid & 31, wave = tid >> 5;
  const int hi = (lane >> 4) & 1, l15 = lane & 15;
  const int mBase = blockIdx.y * 64, nBase = blockIdx.x * 64;
  const int mt = wave & 3, nt0 = (wave >> 2) * 2;

  // copy mapping: each thread moves 2 x 16B per matrix per stage
  const int r0 = tid >> 3;        // 0..31
  const int c0 = (tid & 7) * 8;   // 0..56

  f32x8 acc0, acc1;
  #pragma unroll
  for (int r = 0; r < 8; ++r) { acc0[r] = 0.f; acc1[r] = 0.f; }

  const int nStages = K >> 6;

  auto prefetch = [&](int buf, int k0) {
    async_copy_b128(A + (size_t)(mBase + r0) * K + k0 + c0,      &Asub[buf][r0][c0]);
    async_copy_b128(A + (size_t)(mBase + r0 + 32) * K + k0 + c0, &Asub[buf][r0 + 32][c0]);
    async_copy_b128(B + (size_t)(nBase + r0) * K + k0 + c0,      &Bsub[buf][r0][c0]);
    async_copy_b128(B + (size_t)(nBase + r0 + 32) * K + k0 + c0, &Bsub[buf][r0 + 32][c0]);
  };

  prefetch(0, 0);
  for (int t = 0; t < nStages; ++t) {
    const int cur = t & 1;
    if (t + 1 < nStages) {        // keep next tile's 4 copies in flight
      prefetch(cur ^ 1, (t + 1) << 6);
      wait_async4();
    } else {
      wait_async0();
    }
    __syncthreads();

    #pragma unroll
    for (int ks = 0; ks < 2; ++ks) {
      Frag16 a, b0, b1;
      #pragma unroll
      for (int v = 0; v < 8; ++v) {
        int ka = ks * 32 + ((v >> 2) << 4) + (hi ? 8 : 0) + ((v & 3) << 1);
        a.u[v]  = *(const unsigned int*)&Asub[cur][mt * 16 + l15][ka];
        int kb = ks * 32 + ((v >> 2) << 4) + (hi ? 8 : 0) + ((v & 3) << 1);
        b0.u[v] = *(const unsigned int*)&Bsub[cur][nt0 * 16 + l15][kb];
        b1.u[v] = *(const unsigned int*)&Bsub[cur][(nt0 + 1) * 16 + l15][kb];
      }
      acc0 = wmma_bf16(a, b0, acc0);
      acc1 = wmma_bf16(a, b1, acc1);
    }
    __syncthreads();
  }

  #pragma unroll
  for (int r = 0; r < 8; ++r) {
    int row = mBase + mt * 16 + r + (hi ? 8 : 0);
    size_t i0 = (size_t)row * N + nBase + nt0 * 16 + l15;
    size_t i1 = (size_t)row * N + nBase + (nt0 + 1) * 16 + l15;
    if (mode == 2) {
      Cbf[i0] = f2bf(acc0[r]);
      Cbf[i1] = f2bf(acc1[r]);
    } else {
      float v0 = acc0[r], v1 = acc1[r];
      if (mode == 1) { v0 += Res[i0]; v1 += Res[i1]; }
      Cf[i0] = v0; Cf[i1] = v1;
    }
  }
}

// ---------------------------------------------------------------- gated FFN-up
// Out = bf16( gelu(A@B[0:N]^T) * (A@B[N:2N]^T) ), B: [2N,K]
__global__ __launch_bounds__(256) void gemm_gated_kernel(
    const unsigned short* __restrict__ A, const unsigned short* __restrict__ B,
    unsigned short* __restrict__ Out, int M, int N, int K) {
  __shared__ __align__(16) unsigned short Asub[2][64][64];
  __shared__ __align__(16) unsigned short Usub[2][64][64];
  __shared__ __align__(16) unsigned short Vsub[2][64][64];

  const int tid = threadIdx.x;
  const int lane = tid & 31, wave = tid >> 5;
  const int hi = (lane >> 4) & 1, l15 = lane & 15;
  const int mBase = blockIdx.y * 64, nBase = blockIdx.x * 64;
  const int mt = wave & 3, nt0 = (wave >> 2) * 2;

  const int r0 = tid >> 3;
  const int c0 = (tid & 7) * 8;

  f32x8 au0, au1, av0, av1;
  #pragma unroll
  for (int r = 0; r < 8; ++r) { au0[r] = au1[r] = av0[r] = av1[r] = 0.f; }

  const unsigned short* B2 = B + (size_t)N * K;
  const int nStages = K >> 6;

  auto prefetch = [&](int buf, int k0) {
    async_copy_b128(A  + (size_t)(mBase + r0) * K + k0 + c0,      &Asub[buf][r0][c0]);
    async_copy_b128(A  + (size_t)(mBase + r0 + 32) * K + k0 + c0, &Asub[buf][r0 + 32][c0]);
    async_copy_b128(B  + (size_t)(nBase + r0) * K + k0 + c0,      &Usub[buf][r0][c0]);
    async_copy_b128(B  + (size_t)(nBase + r0 + 32) * K + k0 + c0, &Usub[buf][r0 + 32][c0]);
    async_copy_b128(B2 + (size_t)(nBase + r0) * K + k0 + c0,      &Vsub[buf][r0][c0]);
    async_copy_b128(B2 + (size_t)(nBase + r0 + 32) * K + k0 + c0, &Vsub[buf][r0 + 32][c0]);
  };

  prefetch(0, 0);
  for (int t = 0; t < nStages; ++t) {
    const int cur = t & 1;
    if (t + 1 < nStages) {
      prefetch(cur ^ 1, (t + 1) << 6);
      wait_async6();
    } else {
      wait_async0();
    }
    __syncthreads();

    #pragma unroll
    for (int ks = 0; ks < 2; ++ks) {
      Frag16 a, u0, u1, v0, v1;
      #pragma unroll
      for (int v = 0; v < 8; ++v) {
        int ka = ks * 32 + ((v >> 2) << 4) + (hi ? 8 : 0) + ((v & 3) << 1);
        a.u[v]  = *(const unsigned int*)&Asub[cur][mt * 16 + l15][ka];
        int kb = ks * 32 + ((v >> 2) << 4) + (hi ? 8 : 0) + ((v & 3) << 1);
        u0.u[v] = *(const unsigned int*)&Usub[cur][nt0 * 16 + l15][kb];
        u1.u[v] = *(const unsigned int*)&Usub[cur][(nt0 + 1) * 16 + l15][kb];
        v0.u[v] = *(const unsigned int*)&Vsub[cur][nt0 * 16 + l15][kb];
        v1.u[v] = *(const unsigned int*)&Vsub[cur][(nt0 + 1) * 16 + l15][kb];
      }
      au0 = wmma_bf16(a, u0, au0);
      au1 = wmma_bf16(a, u1, au1);
      av0 = wmma_bf16(a, v0, av0);
      av1 = wmma_bf16(a, v1, av1);
    }
    __syncthreads();
  }

  #pragma unroll
  for (int r = 0; r < 8; ++r) {
    int row = mBase + mt * 16 + r + (hi ? 8 : 0);
    size_t i0 = (size_t)row * N + nBase + nt0 * 16 + l15;
    size_t i1 = (size_t)row * N + nBase + (nt0 + 1) * 16 + l15;
    Out[i0] = f2bf(gelu_exact(au0[r]) * av0[r]);
    Out[i1] = f2bf(gelu_exact(au1[r]) * av1[r]);
  }
}

// ---------------------------------------------------------------- flash attention
// 8 waves/block, each wave owns 16 query rows; streams key blocks of 32.
__global__ __launch_bounds__(256) void attn_kernel(
    const unsigned short* __restrict__ Qb, const unsigned short* __restrict__ Kb,
    const unsigned short* __restrict__ Vb, unsigned short* __restrict__ Zb) {
  __shared__ __align__(16) unsigned short Ks[32][128];   // [key][d]
  __shared__ __align__(16) unsigned short Vst[128][32];  // [d][key]
  __shared__ __align__(16) unsigned short Pw[8][16][32]; // per-wave P tile

  const int tid = threadIdx.x;
  const int lane = tid & 31, wave = tid >> 5;
  const int hi = (lane >> 4) & 1, l15 = lane & 15;

  const int bid = blockIdx.x;
  const int rb = bid & 15;             // T/128 = 16 row blocks
  const int h  = (bid >> 4) & 15;
  const int b  = bid >> 8;
  const int rowBase = rb * 128;
  const int qm0 = rowBase + wave * 16;

  // preload Q fragments (HD=128 -> 4 k-steps of 32)
  Frag16 qf[4];
  {
    size_t qoff = ((size_t)(b * SEQ + qm0 + l15)) * D_MODEL + h * HDIM;
    #pragma unroll
    for (int s = 0; s < 4; ++s)
      #pragma unroll
      for (int v = 0; v < 8; ++v) {
        int kk = s * 32 + ((v >> 2) << 4) + (hi ? 8 : 0) + ((v & 3) << 1);
        qf[s].u[v] = *(const unsigned int*)(Qb + qoff + kk);
      }
  }

  f32x8 o[8];
  #pragma unroll
  for (int nt = 0; nt < 8; ++nt)
    #pragma unroll
    for (int r = 0; r < 8; ++r) o[nt][r] = 0.f;
  float m_i[8], l_i[8];
  #pragma unroll
  for (int r = 0; r < 8; ++r) { m_i[r] = -1e30f; l_i[r] = 0.f; }

  const int nkb = rb * 4 + 4;                    // causal: keys < rowBase+128
  const float qk_scale = 0.08838834764831845f;   // 1/sqrt(128)

  for (int kb = 0; kb < nkb; ++kb) {
    const int keyBase = kb * 32;
    // K tile: async copies straight into LDS (2 x 16B per thread)
    #pragma unroll
    for (int j = 0; j < 2; ++j) {
      int c = tid + j * 256;            // 512 chunks of 8 elems
      int krow = c >> 4, col = (c & 15) * 8;
      async_copy_b128(
          Kb + ((size_t)(b * SEQ + keyBase + krow)) * D_MODEL + h * HDIM + col,
          &Ks[krow][col]);
    }
    // V tile: manual transposed stage (scattered 16-bit stores)
    #pragma unroll
    for (int i = 0; i < 16; ++i) {
      int idx = tid + i * 256;
      int krow = idx >> 7, col = idx & 127;
      size_t g = ((size_t)(b * SEQ + keyBase + krow)) * D_MODEL + h * HDIM + col;
      Vst[col][krow] = Vb[g];
    }
    wait_async0();
    __syncthreads();

    if (keyBase <= qm0 + 15) {
      f32x8 s0, s1;
      #pragma unroll
      for (int r = 0; r < 8; ++r) { s0[r] = 0.f; s1[r] = 0.f; }
      #pragma unroll
      for (int s = 0; s < 4; ++s) {
        Frag16 b0, b1;
        #pragma unroll
        for (int v = 0; v < 8; ++v) {
          int kk = s * 32 + ((v >> 2) << 4) + (hi ? 8 : 0) + ((v & 3) << 1);
          b0.u[v] = *(const unsigned int*)&Ks[l15][kk];
          b1.u[v] = *(const unsigned int*)&Ks[16 + l15][kk];
        }
        s0 = wmma_bf16(qf[s], b0, s0);
        s1 = wmma_bf16(qf[s], b1, s1);
      }

      const int qbase = qm0 + (hi ? 8 : 0);
      float sc[8];
      #pragma unroll
      for (int r = 0; r < 8; ++r) {
        int qr = qbase + r;
        float v0 = s0[r] * qk_scale;
        float v1 = s1[r] * qk_scale;
        if (keyBase + l15 > qr)      v0 = -__builtin_inff();
        if (keyBase + 16 + l15 > qr) v1 = -__builtin_inff();
        float rm = fmaxf(v0, v1);
        #pragma unroll
        for (int off = 8; off > 0; off >>= 1)
          rm = fmaxf(rm, __shfl_xor(rm, off, 32));
        float mn = fmaxf(m_i[r], rm);
        sc[r] = __expf(m_i[r] - mn);
        float p0 = __expf(v0 - mn);
        float p1 = __expf(v1 - mn);
        s0[r] = p0; s1[r] = p1;
        float rs = p0 + p1;
        #pragma unroll
        for (int off = 8; off > 0; off >>= 1)
          rs += __shfl_xor(rs, off, 32);
        l_i[r] = l_i[r] * sc[r] + rs;
        m_i[r] = mn;
      }
      #pragma unroll
      for (int nt = 0; nt < 8; ++nt)
        #pragma unroll
        for (int r = 0; r < 8; ++r) o[nt][r] *= sc[r];

      // re-layout P: C-frag -> LDS -> A-frag (same wave, DS kept in order)
      #pragma unroll
      for (int r = 0; r < 8; ++r) {
        int mrow = r + (hi ? 8 : 0);
        Pw[wave][mrow][l15]      = f2bf(s0[r]);
        Pw[wave][mrow][16 + l15] = f2bf(s1[r]);
      }
      Frag16 pf;
      #pragma unroll
      for (int v = 0; v < 8; ++v) {
        int kk = ((v >> 2) << 4) + (hi ? 8 : 0) + ((v & 3) << 1);
        pf.u[v] = *(const unsigned int*)&Pw[wave][l15][kk];
      }
      #pragma unroll
      for (int nt = 0; nt < 8; ++nt) {
        Frag16 vf;
        #pragma unroll
        for (int v = 0; v < 8; ++v) {
          int kk = ((v >> 2) << 4) + (hi ? 8 : 0) + ((v & 3) << 1);
          vf.u[v] = *(const unsigned int*)&Vst[nt * 16 + l15][kk];
        }
        o[nt] = wmma_bf16(pf, vf, o[nt]);
      }
    }
    __syncthreads();
  }

  #pragma unroll
  for (int nt = 0; nt < 8; ++nt)
    #pragma unroll
    for (int r = 0; r < 8; ++r) {
      int qr = qm0 + r + (hi ? 8 : 0);
      size_t idx = ((size_t)(b * SEQ + qr)) * D_MODEL + h * HDIM + nt * 16 + l15;
      Zb[idx] = f2bf(o[nt][r] / l_i[r]);
    }
}

// ---------------------------------------------------------------- launch
extern "C" void kernel_launch(void* const* d_in, const int* in_sizes, int n_in,
                              void* d_out, int out_size, void* d_ws, size_t ws_size,
                              hipStream_t stream) {
  (void)in_sizes; (void)n_in; (void)out_size; (void)ws_size;
  const float* x   = (const float*)d_in[0];
  const float* g1  = (const float*)d_in[1];
  const float* wq  = (const float*)d_in[2];
  const float* wk  = (const float*)d_in[3];
  const float* wv  = (const float*)d_in[4];
  const float* wo  = (const float*)d_in[5];
  const float* g2  = (const float*)d_in[6];
  const float* wup = (const float*)d_in[7];
  const float* wdn = (const float*)d_in[8];
  float* out = (float*)d_out;

  char* ws = (char*)d_ws;
  size_t off = 0;
  auto alloc = [&](size_t bytes) {
    void* p = ws + off;
    off = (off + bytes + 255) & ~(size_t)255;
    return p;
  };
  unsigned short* wq_bf  = (unsigned short*)alloc((size_t)D_MODEL * D_MODEL * 2);
  unsigned short* wk_bf  = (unsigned short*)alloc((size_t)D_MODEL * D_MODEL * 2);
  unsigned short* wv_bf  = (unsigned short*)alloc((size_t)D_MODEL * D_MODEL * 2);
  unsigned short* wo_bf  = (unsigned short*)alloc((size_t)D_MODEL * D_MODEL * 2);
  unsigned short* wup_bf = (unsigned short*)alloc((size_t)2 * DFF * D_MODEL * 2);
  unsigned short* wdn_bf = (unsigned short*)alloc((size_t)D_MODEL * DFF * 2);
  unsigned short* h_bf   = (unsigned short*)alloc((size_t)BT * D_MODEL * 2);
  unsigned short* q_bf   = (unsigned short*)alloc((size_t)BT * D_MODEL * 2);
  unsigned short* k_bf   = (unsigned short*)alloc((size_t)BT * D_MODEL * 2);
  unsigned short* v_bf   = (unsigned short*)alloc((size_t)BT * D_MODEL * 2);
  unsigned short* z_bf   = (unsigned short*)alloc((size_t)BT * D_MODEL * 2);
  float*          x1     = (float*)alloc((size_t)BT * D_MODEL * 4);
  unsigned short* h2_bf  = (unsigned short*)alloc((size_t)BT * D_MODEL * 2);
  unsigned short* ff_bf  = (unsigned short*)alloc((size_t)BT * DFF * 2);

  const int CVT = 2048;
  f32_to_bf16_kernel<<<CVT, 256, 0, stream>>>(wq,  wq_bf,  (size_t)D_MODEL * D_MODEL);
  f32_to_bf16_kernel<<<CVT, 256, 0, stream>>>(wk,  wk_bf,  (size_t)D_MODEL * D_MODEL);
  f32_to_bf16_kernel<<<CVT, 256, 0, stream>>>(wv,  wv_bf,  (size_t)D_MODEL * D_MODEL);
  f32_to_bf16_kernel<<<CVT, 256, 0, stream>>>(wo,  wo_bf,  (size_t)D_MODEL * D_MODEL);
  f32_to_bf16_kernel<<<CVT, 256, 0, stream>>>(wup, wup_bf, (size_t)2 * DFF * D_MODEL);
  f32_to_bf16_kernel<<<CVT, 256, 0, stream>>>(wdn, wdn_bf, (size_t)D_MODEL * DFF);

  rmsnorm_bf16_kernel<<<BT, 256, 0, stream>>>(x, g1, h_bf);

  dim3 gD(D_MODEL / 64, BT / 64);
  gemm_bf16_kernel<<<gD, 256, 0, stream>>>(h_bf, wq_bf, nullptr, nullptr, q_bf,
                                           BT, D_MODEL, D_MODEL, 2);
  gemm_bf16_kernel<<<gD, 256, 0, stream>>>(h_bf, wk_bf, nullptr, nullptr, k_bf,
                                           BT, D_MODEL, D_MODEL, 2);
  gemm_bf16_kernel<<<gD, 256, 0, stream>>>(h_bf, wv_bf, nullptr, nullptr, v_bf,
                                           BT, D_MODEL, D_MODEL, 2);

  attn_kernel<<<BATCH * NHEADS * (SEQ / 128), 256, 0, stream>>>(q_bf, k_bf, v_bf, z_bf);

  gemm_bf16_kernel<<<gD, 256, 0, stream>>>(z_bf, wo_bf, x, x1, nullptr,
                                           BT, D_MODEL, D_MODEL, 1);

  rmsnorm_bf16_kernel<<<BT, 256, 0, stream>>>(x1, g2, h2_bf);

  dim3 gU(DFF / 64, BT / 64);
  gemm_gated_kernel<<<gU, 256, 0, stream>>>(h2_bf, wup_bf, ff_bf, BT, DFF, D_MODEL);

  gemm_bf16_kernel<<<gD, 256, 0, stream>>>(ff_bf, wdn_bf, x1, out, nullptr,
                                           BT, D_MODEL, DFF, 1);
}
